// AscendDeepseekV2MoE_89601607729826
// MI455X (gfx1250) — matrix operand order repaired and checked by reading, not compiled, not run on hardware
//
#include <hip/hip_runtime.h>

#define T_TOK   2048
#define HIDDEN  1024
#define NEXP    64
#define INTERD  704
#define SHINT   1408
#define TOPK    6
#define NGRP    8
#define TOPG    4
#define MROWS   64
#define MOESCALE 2.5f

typedef __attribute__((ext_vector_type(16))) __bf16 v16bf;
typedef __attribute__((ext_vector_type(8)))  __bf16 v8bf;
typedef __attribute__((ext_vector_type(8)))  float  v8f;

union AFrag { v16bf v; v8bf h[2]; };

__device__ __forceinline__ __bf16 f2bf(float f) { return (__bf16)f; }

__device__ __forceinline__ v8f wmma_bf16(v16bf a, v16bf b, v8f c) {
  // D = A(16x32 bf16) * B(32x16 bf16) + C(16x16 f32)
  return __builtin_amdgcn_wmma_f32_16x16x32_bf16(
      /*neg_a=*/false, a, /*neg_b=*/false, b,
      /*c_mod=*/(short)0, c, /*reuse_a=*/false, /*reuse_b=*/false);
}

// B fragment: lane covers col = lmod, K rows {8h..8h+7, 16+8h..16+8h+7} (h = lhalf)
__device__ __forceinline__ v16bf load_b_frag(const float* __restrict__ W,
                                             int ldw, int krow, int col) {
  v16bf b;
#pragma unroll
  for (int j = 0; j < 16; ++j) b[j] = f2bf(W[(size_t)(krow - (krow & 16) / 2 + j + (j & 8)) * ldw + col]);
  return b;
}

// A fragment from row-major bf16 LDS tile: two 16B chunks per lane
// lanes 0-15: K = {c0..c0+7, c0+16..c0+23}; lanes 16-31: offset by +8 (c0 includes it)
__device__ __forceinline__ v16bf load_a_frag(const __bf16* base, int row, int ld, int c0) {
  AFrag f;
  const __bf16* p = base + (size_t)row * ld + c0;
  f.h[0] = *(const v8bf*)(p);
  f.h[1] = *(const v8bf*)(p + 16);
  return f.v;
}

// ---------------- Router: logits + grouped top-k + per-expert token lists ---
__global__ __launch_bounds__(64) void moe_router(
    const float* __restrict__ x, const float* __restrict__ gw,
    const float* __restrict__ bias, int* __restrict__ cnt,
    int* __restrict__ etok, float* __restrict__ ew) {
  __shared__ float lg[NEXP];
  const int t = blockIdx.x, e = threadIdx.x;
  const float4* xr = (const float4*)(x + (size_t)t * HIDDEN);
  const float4* wr = (const float4*)(gw + (size_t)e * HIDDEN);
  float s = 0.f;
#pragma unroll 4
  for (int k = 0; k < HIDDEN / 4; ++k) {
    float4 a = xr[k], b = wr[k];
    s += a.x * b.x + a.y * b.y + a.z * b.z + a.w * b.w;
  }
  lg[e] = s;
  __syncthreads();
  if (e == 0) {
    float sig[NEXP], sc[NEXP];
    for (int i = 0; i < NEXP; ++i) {
      sig[i] = 1.f / (1.f + __expf(-lg[i]));
      sc[i] = sig[i] + bias[i];
    }
    // group score = sum of top-2 biased scores in each group of 8
    float gsc[NGRP];
    for (int g = 0; g < NGRP; ++g) {
      float m1 = -1e30f, m2 = -1e30f;
      for (int j = 0; j < NEXP / NGRP; ++j) {
        float v = sc[g * (NEXP / NGRP) + j];
        if (v > m1) { m2 = m1; m1 = v; } else if (v > m2) { m2 = v; }
      }
      gsc[g] = m1 + m2;
    }
    bool keep[NGRP];
    for (int g = 0; g < NGRP; ++g) keep[g] = false;
    for (int it = 0; it < TOPG; ++it) {
      int bi = 0; float bv = -1e30f;
      for (int g = 0; g < NGRP; ++g)
        if (!keep[g] && gsc[g] > bv) { bv = gsc[g]; bi = g; }
      keep[bi] = true;
    }
    float mk[NEXP];
    for (int i = 0; i < NEXP; ++i)
      mk[i] = keep[i / (NEXP / NGRP)] ? sc[i] : -1e30f;
    int ids[TOPK]; float w[TOPK]; float sum = 0.f;
    for (int k = 0; k < TOPK; ++k) {
      int bi = 0; float bv = -3e38f;
      for (int i = 0; i < NEXP; ++i)
        if (mk[i] > bv) { bv = mk[i]; bi = i; }
      mk[bi] = -3e38f;
      ids[k] = bi; w[k] = sig[bi]; sum += w[k];
    }
    float inv = 1.f / sum;
    for (int k = 0; k < TOPK; ++k) {
      int slot = atomicAdd(&cnt[ids[k]], 1);
      etok[ids[k] * T_TOK + slot] = t;
      ew[ids[k] * T_TOK + slot] = w[k] * inv;
    }
  }
}

// ---------------- Routed experts: gather -> (silu(xWg)*xWu) Wd -> scatter ----
__global__ __launch_bounds__(128) void moe_experts(
    const float* __restrict__ x, const float* __restrict__ wg_all,
    const float* __restrict__ wu_all, const float* __restrict__ wd_all,
    const int* __restrict__ cnt, const int* __restrict__ etok,
    const float* __restrict__ ew, float* __restrict__ out) {
  extern __shared__ __align__(16) char smem[];
  __bf16* A  = (__bf16*)smem;                 // 64 x 1024 bf16   = 128 KB
  __bf16* H  = A + MROWS * HIDDEN;            // 64 x  704 bf16   =  88 KB
  float*  cw = (float*)(H + MROWS * INTERD);  // 64
  int*    tok = (int*)(cw + MROWS);           // 64

  const int e = blockIdx.x;
  const int chunk = blockIdx.y;
  const int n = cnt[e];
  const int base = chunk * MROWS;
  if (base >= n) return;
  const int rows = min(MROWS, n - base);

  const int tid = threadIdx.x;
  for (int i = tid; i < MROWS; i += 128) {
    if (i < rows) {
      tok[i] = etok[e * T_TOK + base + i];
      cw[i]  = ew[e * T_TOK + base + i] * MOESCALE;
    } else { tok[i] = 0; cw[i] = 0.f; }
  }
  __syncthreads();
  // gather tokens, convert fp32 -> bf16 into LDS A (zero pad)
  for (int i = tid; i < MROWS * (HIDDEN / 4); i += 128) {
    int r = i >> 8;            // HIDDEN/4 == 256
    int c = (i & 255) << 2;
    float4 v = make_float4(0.f, 0.f, 0.f, 0.f);
    if (r < rows) v = *(const float4*)(x + (size_t)tok[r] * HIDDEN + c);
    A[r * HIDDEN + c + 0] = f2bf(v.x);
    A[r * HIDDEN + c + 1] = f2bf(v.y);
    A[r * HIDDEN + c + 2] = f2bf(v.z);
    A[r * HIDDEN + c + 3] = f2bf(v.w);
  }
  __syncthreads();

  const int wave = tid >> 5, lane = tid & 31;
  const int lhalf = lane >> 4, lmod = lane & 15;
  const float* Wg = wg_all + (size_t)e * HIDDEN * INTERD;
  const float* Wu = wu_all + (size_t)e * HIDDEN * INTERD;
  const v8f vzero = {0.f, 0.f, 0.f, 0.f, 0.f, 0.f, 0.f, 0.f};

  // Stage 1: H[64 x 704] = silu(A@Wg) * (A@Wu), 44 n-tiles over 4 waves
  for (int nt = wave; nt < INTERD / 16; nt += 4) {
    const int col = nt * 16 + lmod;
    v8f accg[4], accu[4];
#pragma unroll
    for (int mt = 0; mt < 4; ++mt) { accg[mt] = vzero; accu[mt] = vzero; }
    for (int ks = 0; ks < HIDDEN / 32; ++ks) {
      const int kb = ks * 32;
      const int krow = kb + lhalf * 16;
      if (ks + 1 < HIDDEN / 32) {
        __builtin_prefetch(&Wg[(size_t)(krow + 32) * INTERD + col], 0, 1);
        __builtin_prefetch(&Wu[(size_t)(krow + 32) * INTERD + col], 0, 1);
      }
      v16bf bg = load_b_frag(Wg, INTERD, krow, col);
      v16bf bu = load_b_frag(Wu, INTERD, krow, col);
      const int c0 = kb + lhalf * 8;
#pragma unroll
      for (int mt = 0; mt < 4; ++mt) {
        v16bf a = load_a_frag(A, mt * 16 + lmod, HIDDEN, c0);
        accg[mt] = wmma_bf16(a, bg, accg[mt]);
        accu[mt] = wmma_bf16(a, bu, accu[mt]);
      }
    }
#pragma unroll
    for (int mt = 0; mt < 4; ++mt)
#pragma unroll
      for (int v = 0; v < 8; ++v) {
        int r = mt * 16 + v + lhalf * 8;
        float g = accg[mt][v], u = accu[mt][v];
        float h = (g / (1.f + __expf(-g))) * u;
        H[r * INTERD + col] = f2bf(h);
      }
  }
  __syncthreads();

  // Stage 2: out += cw * (H @ Wd), 64 n-tiles over 4 waves, K = 704
  const float* Wd = wd_all + (size_t)e * INTERD * HIDDEN;
  for (int nt = wave; nt < HIDDEN / 16; nt += 4) {
    const int col = nt * 16 + lmod;
    v8f acc[4];
#pragma unroll
    for (int mt = 0; mt < 4; ++mt) acc[mt] = vzero;
    for (int ks = 0; ks < INTERD / 32; ++ks) {
      const int kb = ks * 32;
      const int krow = kb + lhalf * 16;
      if (ks + 1 < INTERD / 32)
        __builtin_prefetch(&Wd[(size_t)(krow + 32) * HIDDEN + col], 0, 1);
      v16bf b = load_b_frag(Wd, HIDDEN, krow, col);
      const int c0 = kb + lhalf * 8;
#pragma unroll
      for (int mt = 0; mt < 4; ++mt) {
        v16bf a = load_a_frag(H, mt * 16 + lmod, INTERD, c0);
        acc[mt] = wmma_bf16(a, b, acc[mt]);
      }
    }
#pragma unroll
    for (int mt = 0; mt < 4; ++mt)
#pragma unroll
      for (int v = 0; v < 8; ++v) {
        int r = mt * 16 + v + lhalf * 8;
        if (r < rows)
          atomicAdd(&out[(size_t)tok[r] * HIDDEN + col], acc[mt][v] * cw[r]);
      }
  }
}

// ---------------- Shared experts: dense, INTER = 1408 ----------------------
__global__ __launch_bounds__(128) void moe_shared(
    const float* __restrict__ x, const float* __restrict__ sg,
    const float* __restrict__ su, const float* __restrict__ sd,
    float* __restrict__ out) {
  extern __shared__ __align__(16) char smem[];
  __bf16* A = (__bf16*)smem;           // 64 x 1024 bf16 = 128 KB
  __bf16* H = A + MROWS * HIDDEN;      // 64 x 1408 bf16 = 176 KB

  const int base = blockIdx.x * MROWS;
  const int tid = threadIdx.x;
  for (int i = tid; i < MROWS * (HIDDEN / 4); i += 128) {
    int r = i >> 8;
    int c = (i & 255) << 2;
    float4 v = *(const float4*)(x + (size_t)(base + r) * HIDDEN + c);
    A[r * HIDDEN + c + 0] = f2bf(v.x);
    A[r * HIDDEN + c + 1] = f2bf(v.y);
    A[r * HIDDEN + c + 2] = f2bf(v.z);
    A[r * HIDDEN + c + 3] = f2bf(v.w);
  }
  __syncthreads();

  const int wave = tid >> 5, lane = tid & 31;
  const int lhalf = lane >> 4, lmod = lane & 15;
  const v8f vzero = {0.f, 0.f, 0.f, 0.f, 0.f, 0.f, 0.f, 0.f};

  for (int nt = wave; nt < SHINT / 16; nt += 4) {   // 88 n-tiles
    const int col = nt * 16 + lmod;
    v8f accg[4], accu[4];
#pragma unroll
    for (int mt = 0; mt < 4; ++mt) { accg[mt] = vzero; accu[mt] = vzero; }
    for (int ks = 0; ks < HIDDEN / 32; ++ks) {
      const int kb = ks * 32;
      const int krow = kb + lhalf * 16;
      if (ks + 1 < HIDDEN / 32) {
        __builtin_prefetch(&sg[(size_t)(krow + 32) * SHINT + col], 0, 1);
        __builtin_prefetch(&su[(size_t)(krow + 32) * SHINT + col], 0, 1);
      }
      v16bf bg = load_b_frag(sg, SHINT, krow, col);
      v16bf bu = load_b_frag(su, SHINT, krow, col);
      const int c0 = kb + lhalf * 8;
#pragma unroll
      for (int mt = 0; mt < 4; ++mt) {
        v16bf a = load_a_frag(A, mt * 16 + lmod, HIDDEN, c0);
        accg[mt] = wmma_bf16(a, bg, accg[mt]);
        accu[mt] = wmma_bf16(a, bu, accu[mt]);
      }
    }
#pragma unroll
    for (int mt = 0; mt < 4; ++mt)
#pragma unroll
      for (int v = 0; v < 8; ++v) {
        int r = mt * 16 + v + lhalf * 8;
        float g = accg[mt][v], u = accu[mt][v];
        float h = (g / (1.f + __expf(-g))) * u;
        H[r * SHINT + col] = f2bf(h);
      }
  }
  __syncthreads();

  for (int nt = wave; nt < HIDDEN / 16; nt += 4) {
    const int col = nt * 16 + lmod;
    v8f acc[4];
#pragma unroll
    for (int mt = 0; mt < 4; ++mt) acc[mt] = vzero;
    for (int ks = 0; ks < SHINT / 32; ++ks) {       // 44 k-steps
      const int kb = ks * 32;
      const int krow = kb + lhalf * 16;
      if (ks + 1 < SHINT / 32)
        __builtin_prefetch(&sd[(size_t)(krow + 32) * HIDDEN + col], 0, 1);
      v16bf b = load_b_frag(sd, HIDDEN, krow, col);
      const int c0 = kb + lhalf * 8;
#pragma unroll
      for (int mt = 0; mt < 4; ++mt) {
        v16bf a = load_a_frag(H, mt * 16 + lmod, SHINT, c0);
        acc[mt] = wmma_bf16(a, b, acc[mt]);
      }
    }
#pragma unroll
    for (int mt = 0; mt < 4; ++mt)
#pragma unroll
      for (int v = 0; v < 8; ++v) {
        int r = mt * 16 + v + lhalf * 8;
        atomicAdd(&out[(size_t)(base + r) * HIDDEN + col], acc[mt][v]);
      }
  }
}

extern "C" void kernel_launch(void* const* d_in, const int* in_sizes, int n_in,
                              void* d_out, int out_size, void* d_ws, size_t ws_size,
                              hipStream_t stream) {
  const float* x    = (const float*)d_in[0];
  const float* gw   = (const float*)d_in[1];
  const float* bias = (const float*)d_in[2];
  const float* wg   = (const float*)d_in[3];
  const float* wu   = (const float*)d_in[4];
  const float* wd   = (const float*)d_in[5];
  const float* sg   = (const float*)d_in[6];
  const float* su   = (const float*)d_in[7];
  const float* sd   = (const float*)d_in[8];
  float* out = (float*)d_out;

  char* ws = (char*)d_ws;
  int*   cnt  = (int*)ws;                                        // 64 ints
  int*   etok = (int*)(ws + 256);                                // 64*2048 ints
  float* ew   = (float*)(ws + 256 + (size_t)NEXP * T_TOK * 4);   // 64*2048 floats

  hipMemsetAsync(cnt, 0, 256, stream);
  hipMemsetAsync(d_out, 0, (size_t)out_size * sizeof(float), stream);

  moe_router<<<T_TOK, NEXP, 0, stream>>>(x, gw, bias, cnt, etok, ew);

  const size_t lds_e = (size_t)MROWS * HIDDEN * 2 + (size_t)MROWS * INTERD * 2
                     + (size_t)MROWS * 8;                        // 221,696 B
  const size_t lds_s = (size_t)MROWS * HIDDEN * 2 + (size_t)MROWS * SHINT * 2; // 311,296 B
  hipFuncSetAttribute((const void*)moe_experts,
                      hipFuncAttributeMaxDynamicSharedMemorySize, (int)lds_e);
  hipFuncSetAttribute((const void*)moe_shared,
                      hipFuncAttributeMaxDynamicSharedMemorySize, (int)lds_s);

  moe_experts<<<dim3(NEXP, T_TOK / MROWS), 128, lds_e, stream>>>(
      x, wg, wu, wd, cnt, etok, ew, out);
  moe_shared<<<T_TOK / MROWS, 128, lds_s, stream>>>(x, sg, su, sd, out);
}
